// cswinTransformerModel_87651692577015
// MI455X (gfx1250) — matrix operand order repaired and checked by reading, not compiled, not run on hardware
//
#include <hip/hip_runtime.h>
#include <hip/hip_bf16.h>

typedef __bf16 bf16;
typedef __bf16 v16bf __attribute__((ext_vector_type(16)));
typedef float  v8f   __attribute__((ext_vector_type(8)));
typedef int    v4i   __attribute__((ext_vector_type(4)));

#define WMMA_BF16(a, b, c) \
  __builtin_amdgcn_wmma_f32_16x16x32_bf16(false, (a), false, (b), (short)0, (c), false, false)

#if defined(__has_builtin)
#if __has_builtin(__builtin_amdgcn_global_load_async_to_lds_b128)
#define ASYNC_LDS 1
#else
#define ASYNC_LDS 0
#endif
#else
#define ASYNC_LDS 0
#endif

typedef __attribute__((address_space(1))) v4i gv4i;  // global int4
typedef __attribute__((address_space(3))) v4i lv4i;  // LDS int4

__device__ __forceinline__ gv4i* as_gbl4(const void* p) {
  // generic global pointer bits == 64-bit VA
  return (gv4i*)(unsigned long long)p;
}
__device__ __forceinline__ lv4i* as_lds4(void* p) {
  // generic LDS pointer: low 32 bits are the LDS byte offset (ISA aperture rule)
  return (lv4i*)(unsigned int)(unsigned long long)p;
}
__device__ __forceinline__ void wait_async0() {
#if defined(__has_builtin) && __has_builtin(__builtin_amdgcn_s_wait_asynccnt)
  __builtin_amdgcn_s_wait_asynccnt(0);
#else
  asm volatile("s_wait_asynccnt 0x0" ::: "memory");
#endif
}

__device__ __forceinline__ v8f zero8() {
  v8f z = {0.f, 0.f, 0.f, 0.f, 0.f, 0.f, 0.f, 0.f};
  return z;
}

// A-fragment (16x32 bf16, M x K): lane L -> row L%16; element i -> K = i + 8*(i>=8) + 8*(L>=16)
__device__ __forceinline__ v16bf frag_a(const bf16* base, int stride, int lane) {
  const bf16* p = base + (lane & 15) * stride + ((lane >> 4) << 3);
  union { uint4 q[2]; v16bf v; } u;
  u.q[0] = *(const uint4*)(p);
  u.q[1] = *(const uint4*)(p + 16);
  return u.v;
}

// B-fragment (32x16 bf16, K x N) from [N][K] storage: lane L -> col L%16; element i -> K = i + 8*(i>=8) + 8*(L>=16)
__device__ __forceinline__ v16bf frag_b(const bf16* base, int stride, int lane) {
  const bf16* p = base + (lane & 15) * stride + ((lane >> 4) << 3);
  union { uint4 q[2]; v16bf v; } u;
  u.q[0] = *(const uint4*)(p);
  u.q[1] = *(const uint4*)(p + 16);
  return u.v;
}

// ---------------------------------------------------------------------------
// float -> bf16 conversion (weights)
// ---------------------------------------------------------------------------
__global__ void cvt_bf16_kernel(const float* __restrict__ src, bf16* __restrict__ dst, int n) {
  int i = blockIdx.x * 256 + threadIdx.x;
  if (i < n) dst[i] = (bf16)src[i];
}

// ---------------------------------------------------------------------------
// LayerNorm over C=512, one wave per row, output bf16
// ---------------------------------------------------------------------------
__global__ __launch_bounds__(256) void ln_bf16_kernel(const float* __restrict__ x,
                                                      const float* __restrict__ gam,
                                                      const float* __restrict__ bet,
                                                      bf16* __restrict__ y) {
  const int lane = threadIdx.x & 31;
  const int row  = blockIdx.x * 8 + (threadIdx.x >> 5);
  const float* xr = x + (size_t)row * 512;
  float4 v[4];
  float s = 0.f;
#pragma unroll
  for (int j = 0; j < 4; ++j) {
    v[j] = *(const float4*)(xr + lane * 4 + 128 * j);
    s += v[j].x + v[j].y + v[j].z + v[j].w;
  }
#pragma unroll
  for (int m = 1; m < 32; m <<= 1) s += __shfl_xor(s, m, 32);
  const float mean = s * (1.0f / 512.0f);
  float q = 0.f;
#pragma unroll
  for (int j = 0; j < 4; ++j) {
    float a = v[j].x - mean, b2 = v[j].y - mean, c = v[j].z - mean, d = v[j].w - mean;
    q += a * a + b2 * b2 + c * c + d * d;
  }
#pragma unroll
  for (int m = 1; m < 32; m <<= 1) q += __shfl_xor(q, m, 32);
  const float rstd = rsqrtf(q * (1.0f / 512.0f) + 1e-5f);
  bf16* yr = y + (size_t)row * 512;
#pragma unroll
  for (int j = 0; j < 4; ++j) {
    const int c0 = lane * 4 + 128 * j;
    const float* vv = &v[j].x;
#pragma unroll
    for (int i = 0; i < 4; ++i) {
      float o = (vv[i] - mean) * rstd * gam[c0 + i] + bet[c0 + i];
      yr[c0 + i] = (bf16)o;
    }
  }
}

// ---------------------------------------------------------------------------
// GEMM: out[M,N] = A[M,K](bf16) @ W[N,K]^T (bf16) (+bias)(+epilogue)
// EPI: 0 -> store bf16; 1 -> bias+exact GELU -> bf16; 2 -> bias + residual -> f32
// block = 128(M) x 256(N) tile, 8 waves (2x4) of 64x64 each,
// K chunk = 32, double-buffered LDS, one barrier per chunk.
// Staging uses GLOBAL_LOAD_ASYNC_TO_LDS_B128 (ASYNCcnt) when available.
// ---------------------------------------------------------------------------
template <int EPI>
__global__ __launch_bounds__(256) void gemm_kernel(const bf16* __restrict__ A, int lda,
                                                   const bf16* __restrict__ W, int K,
                                                   const float* __restrict__ bias,
                                                   const float* __restrict__ resid,
                                                   float* __restrict__ outf,
                                                   bf16* __restrict__ outb, int ldo) {
  __shared__ bf16 As[2][128][40];
  __shared__ bf16 Bs[2][256][40];
  const int tid = threadIdx.x;
  const int lane = tid & 31, wid = tid >> 5;
  const int wm = wid & 1, wn = wid >> 1;  // wave tile at rows wm*64, cols wn*64
  const int m0 = blockIdx.y * 128, n0 = blockIdx.x * 256;

  v8f acc[4][4];
#pragma unroll
  for (int i = 0; i < 4; ++i)
#pragma unroll
    for (int j = 0; j < 4; ++j) acc[i][j] = zero8();

  // staging: A 128x32 -> thread t: row t>>1, cols (t&1)*16..+15 (2 x b128)
  //          B 256x32 -> thread t: row t,    cols 0..31        (4 x b128)
  const int ar = tid >> 1, ac = (tid & 1) * 16;
  const bf16* gA = A + (size_t)(m0 + ar) * lda + ac;
  const bf16* gB = W + (size_t)(n0 + tid) * K;
  const int nk = K >> 5;

#if ASYNC_LDS
  auto stage_async = [&](int p, int k0) {
    __builtin_amdgcn_global_load_async_to_lds_b128(as_gbl4(gA + k0),      as_lds4(&As[p][ar][ac]),     0, 0);
    __builtin_amdgcn_global_load_async_to_lds_b128(as_gbl4(gA + k0 + 8),  as_lds4(&As[p][ar][ac + 8]), 0, 0);
    __builtin_amdgcn_global_load_async_to_lds_b128(as_gbl4(gB + k0),      as_lds4(&Bs[p][tid][0]),     0, 0);
    __builtin_amdgcn_global_load_async_to_lds_b128(as_gbl4(gB + k0 + 8),  as_lds4(&Bs[p][tid][8]),     0, 0);
    __builtin_amdgcn_global_load_async_to_lds_b128(as_gbl4(gB + k0 + 16), as_lds4(&Bs[p][tid][16]),    0, 0);
    __builtin_amdgcn_global_load_async_to_lds_b128(as_gbl4(gB + k0 + 24), as_lds4(&Bs[p][tid][24]),    0, 0);
  };
  stage_async(0, 0);
  wait_async0();
  __syncthreads();
#else
  {
    uint4 a0 = *(const uint4*)(gA);
    uint4 a1 = *(const uint4*)(gA + 8);
    uint4 b0 = *(const uint4*)(gB);
    uint4 b1 = *(const uint4*)(gB + 8);
    uint4 b2 = *(const uint4*)(gB + 16);
    uint4 b3 = *(const uint4*)(gB + 24);
    *(uint4*)&As[0][ar][ac]     = a0;
    *(uint4*)&As[0][ar][ac + 8] = a1;
    *(uint4*)&Bs[0][tid][0]  = b0;
    *(uint4*)&Bs[0][tid][8]  = b1;
    *(uint4*)&Bs[0][tid][16] = b2;
    *(uint4*)&Bs[0][tid][24] = b3;
  }
  __syncthreads();
#endif

  for (int kc = 0; kc < nk; ++kc) {
    const int p = kc & 1;
    const bool more = (kc + 1) < nk;
#if ASYNC_LDS
    if (more) stage_async(p ^ 1, (kc + 1) << 5);  // async engine fills other buffer
#else
    uint4 a0, a1, b0, b1, b2, b3;
    if (more) {
      const int k0 = (kc + 1) << 5;
      a0 = *(const uint4*)(gA + k0);
      a1 = *(const uint4*)(gA + k0 + 8);
      b0 = *(const uint4*)(gB + k0);
      b1 = *(const uint4*)(gB + k0 + 8);
      b2 = *(const uint4*)(gB + k0 + 16);
      b3 = *(const uint4*)(gB + k0 + 24);
    }
#endif
    v16bf af[4];
#pragma unroll
    for (int i = 0; i < 4; ++i) af[i] = frag_a(&As[p][wm * 64 + i * 16][0], 40, lane);
#pragma unroll
    for (int j = 0; j < 4; ++j) {
      const v16bf bfr = frag_b(&Bs[p][wn * 64 + j * 16][0], 40, lane);
#pragma unroll
      for (int i = 0; i < 4; ++i) acc[i][j] = WMMA_BF16(af[i], bfr, acc[i][j]);
    }
#if ASYNC_LDS
    wait_async0();  // our async copies into buffer p^1 have landed in LDS
#else
    if (more) {
      *(uint4*)&As[p ^ 1][ar][ac]     = a0;
      *(uint4*)&As[p ^ 1][ar][ac + 8] = a1;
      *(uint4*)&Bs[p ^ 1][tid][0]  = b0;
      *(uint4*)&Bs[p ^ 1][tid][8]  = b1;
      *(uint4*)&Bs[p ^ 1][tid][16] = b2;
      *(uint4*)&Bs[p ^ 1][tid][24] = b3;
    }
#endif
    __syncthreads();
  }

#pragma unroll
  for (int i = 0; i < 4; ++i) {
#pragma unroll
    for (int j = 0; j < 4; ++j) {
      const int colb = n0 + wn * 64 + j * 16 + (lane & 15);
      const float bia = bias ? bias[colb] : 0.f;
#pragma unroll
      for (int jj = 0; jj < 8; ++jj) {
        const int row = m0 + wm * 64 + i * 16 + jj + ((lane >> 4) << 3);
        float vv = acc[i][j][jj] + bia;
        if (EPI == 2) {
          outf[(size_t)row * ldo + colb] = resid[(size_t)row * ldo + colb] + vv;
        } else {
          if (EPI == 1) vv = 0.5f * vv * (1.0f + erff(vv * 0.70710678118654752f));
          outb[(size_t)row * ldo + colb] = (bf16)vv;
        }
      }
    }
  }
}

// ---------------------------------------------------------------------------
// Strip attention. qkv: [B*N, 1536] bf16 (q|k|v, each head h at col h*64).
// Every stripe = contiguous 512 tokens (both h/v branches are flat reshapes).
// Block = (stripe, 128-query block); 8 waves x 16 query rows, flash softmax
// over 8 key chunks of 64. Output attn[B*N, 512] bf16, head h at col h*64.
// ---------------------------------------------------------------------------
__global__ __launch_bounds__(256) void attn_kernel(const bf16* __restrict__ qkv,
                                                   bf16* __restrict__ attn) {
  __shared__ bf16 Qs[128][72];
  __shared__ bf16 Ks[64][72];      // [key][d]  -> B side of Q*K^T
  __shared__ bf16 Vt[64][72];      // [d][key]  -> B side of P*V
  __shared__ bf16 Ps[8][16][72];   // per-wave P scratch

  const int tid = threadIdx.x, lane = tid & 31, wid = tid >> 5;
  const int s = blockIdx.x, mb = blockIdx.y;
  const int b = s >> 6, h = (s >> 3) & 7, g = s & 7;
  const bf16* base = qkv + (size_t)(b * 4096 + g * 512) * 1536 + h * 64;

  {  // stage 128 query rows x 64 dims
    const int r = tid >> 1, c = (tid & 1) * 32;
    const bf16* gq = base + (size_t)(mb * 128 + r) * 1536 + c;
    *(uint4*)&Qs[r][c]      = *(const uint4*)(gq);
    *(uint4*)&Qs[r][c + 8]  = *(const uint4*)(gq + 8);
    *(uint4*)&Qs[r][c + 16] = *(const uint4*)(gq + 16);
    *(uint4*)&Qs[r][c + 24] = *(const uint4*)(gq + 24);
  }
  __syncthreads();
  const v16bf qa0 = frag_a(&Qs[wid * 16][0], 72, lane);
  const v16bf qa1 = frag_a(&Qs[wid * 16][32], 72, lane);

  v8f o[4];
  float runmax[8], runsum[8];
#pragma unroll
  for (int t = 0; t < 4; ++t) o[t] = zero8();
#pragma unroll
  for (int j = 0; j < 8; ++j) { runmax[j] = -1e30f; runsum[j] = 0.f; }
  const float scale = 0.125f;  // HD^-0.5

  for (int kc = 0; kc < 8; ++kc) {
    __syncthreads();
    {  // stage K chunk (as [key][d]) and V chunk transposed (as [d][key])
      const int r = tid >> 2, c = (tid & 3) * 16;
      const bf16* gk = base + 512 + (size_t)(kc * 64 + r) * 1536 + c;
      const bf16* gv = base + 1024 + (size_t)(kc * 64 + r) * 1536 + c;
      *(uint4*)&Ks[r][c]     = *(const uint4*)(gk);
      *(uint4*)&Ks[r][c + 8] = *(const uint4*)(gk + 8);
      union { uint4 q[2]; bf16 e[16]; } tv;
      tv.q[0] = *(const uint4*)(gv);
      tv.q[1] = *(const uint4*)(gv + 8);
#pragma unroll
      for (int i = 0; i < 16; ++i) Vt[c + i][r] = tv.e[i];
    }
    __syncthreads();

    v8f sv[4];
#pragma unroll
    for (int nt = 0; nt < 4; ++nt) {
      v16bf kb0 = frag_b(&Ks[nt * 16][0], 72, lane);
      v16bf kb1 = frag_b(&Ks[nt * 16][32], 72, lane);
      v8f z = zero8();
      z = WMMA_BF16(qa0, kb0, z);
      sv[nt] = WMMA_BF16(qa1, kb1, z);
    }

    // online softmax: row r (r = jj + 8*(lane>=16)) lives on a 16-lane group
#pragma unroll
    for (int jj = 0; jj < 8; ++jj) {
      float p4[4];
#pragma unroll
      for (int nt = 0; nt < 4; ++nt) p4[nt] = sv[nt][jj] * scale;
      float m = fmaxf(fmaxf(p4[0], p4[1]), fmaxf(p4[2], p4[3]));
#pragma unroll
      for (int k = 1; k < 16; k <<= 1) m = fmaxf(m, __shfl_xor(m, k, 32));
      const float nm = fmaxf(runmax[jj], m);
      const float cf = __expf(runmax[jj] - nm);
      runmax[jj] = nm;
      float ps = 0.f;
#pragma unroll
      for (int nt = 0; nt < 4; ++nt) {
        const float e = __expf(p4[nt] - nm);
        ps += e;
        Ps[wid][jj + ((lane >> 4) << 3)][nt * 16 + (lane & 15)] = (bf16)e;
      }
#pragma unroll
      for (int k = 1; k < 16; k <<= 1) ps += __shfl_xor(ps, k, 32);
      runsum[jj] = runsum[jj] * cf + ps;
#pragma unroll
      for (int t = 0; t < 4; ++t) o[t][jj] *= cf;
    }

    // O += P @ Vc   (per-wave Ps: same-wave LDS RAW, in-order via DScnt)
    const v16bf pa0 = frag_a(&Ps[wid][0][0], 72, lane);
    const v16bf pa1 = frag_a(&Ps[wid][0][32], 72, lane);
#pragma unroll
    for (int dt = 0; dt < 4; ++dt) {
      v16bf vb0 = frag_b(&Vt[dt * 16][0], 72, lane);
      v16bf vb1 = frag_b(&Vt[dt * 16][32], 72, lane);
      o[dt] = WMMA_BF16(pa0, vb0, o[dt]);
      o[dt] = WMMA_BF16(pa1, vb1, o[dt]);
    }
  }

  const int n0 = g * 512 + mb * 128 + wid * 16 + ((lane >> 4) << 3);
  bf16* ob = attn + (size_t)(b * 4096 + n0) * 512 + h * 64 + (lane & 15);
#pragma unroll
  for (int jj = 0; jj < 8; ++jj) {
    const float inv = 1.0f / runsum[jj];
#pragma unroll
    for (int dt = 0; dt < 4; ++dt) ob[(size_t)jj * 512 + dt * 16] = (bf16)(o[dt][jj] * inv);
  }
}

// ---------------------------------------------------------------------------
extern "C" void kernel_launch(void* const* d_in, const int* in_sizes, int n_in,
                              void* d_out, int out_size, void* d_ws, size_t ws_size,
                              hipStream_t stream) {
  (void)in_sizes; (void)n_in; (void)out_size; (void)ws_size;
  const float* x      = (const float*)d_in[0];
  const float* ln1_g  = (const float*)d_in[1];
  const float* ln1_b  = (const float*)d_in[2];
  const float* qkv_w  = (const float*)d_in[3];
  const float* proj_w = (const float*)d_in[4];
  const float* proj_b = (const float*)d_in[5];
  const float* ln2_g  = (const float*)d_in[6];
  const float* ln2_b  = (const float*)d_in[7];
  const float* ff1_w  = (const float*)d_in[8];
  const float* ff1_b  = (const float*)d_in[9];
  const float* ff2_w  = (const float*)d_in[10];
  const float* ff2_b  = (const float*)d_in[11];

  const int M = 8 * 4096;  // 32768 tokens
  char* w = (char*)d_ws;
  float* xs = (float*)w;  w += (size_t)M * 512 * 4;       // residual stream (f32)
  bf16* ybf = (bf16*)w;   w += (size_t)M * 512 * 2;       // LN out / attn out (bf16)
  bf16* big = (bf16*)w;   w += (size_t)M * 2048 * 2;      // qkv (ld 1536) / ffn hidden (ld 2048)
  bf16* wq  = (bf16*)w;   w += (size_t)2 * 1536 * 512 * 2;
  bf16* wp  = (bf16*)w;   w += (size_t)2 * 512 * 512 * 2;
  bf16* w1  = (bf16*)w;   w += (size_t)2 * 2048 * 512 * 2;
  bf16* w2  = (bf16*)w;   w += (size_t)2 * 512 * 2048 * 2;

  cvt_bf16_kernel<<<(2 * 1536 * 512 + 255) / 256, 256, 0, stream>>>(qkv_w, wq, 2 * 1536 * 512);
  cvt_bf16_kernel<<<(2 * 512 * 512 + 255) / 256, 256, 0, stream>>>(proj_w, wp, 2 * 512 * 512);
  cvt_bf16_kernel<<<(2 * 2048 * 512 + 255) / 256, 256, 0, stream>>>(ff1_w, w1, 2 * 2048 * 512);
  cvt_bf16_kernel<<<(2 * 512 * 2048 + 255) / 256, 256, 0, stream>>>(ff2_w, w2, 2 * 512 * 2048);

  for (int i = 0; i < 2; ++i) {
    const float* R = (i == 0) ? x : xs;
    // y = LN1(R)
    ln_bf16_kernel<<<4096, 256, 0, stream>>>(R, ln1_g + i * 512, ln1_b + i * 512, ybf);
    // qkv = y @ qkv_w^T  (no bias)
    gemm_kernel<0><<<dim3(6, 256), 256, 0, stream>>>(ybf, 512, wq + (size_t)i * 1536 * 512, 512,
                                                     nullptr, nullptr, nullptr, big, 1536);
    // strip attention (attn out overwrites ybf)
    attn_kernel<<<dim3(512, 4), 256, 0, stream>>>(big, ybf);
    // xs = R + attn @ proj_w^T + proj_b
    gemm_kernel<2><<<dim3(2, 256), 256, 0, stream>>>(ybf, 512, wp + (size_t)i * 512 * 512, 512,
                                                     proj_b + i * 512, R, xs, nullptr, 512);
    // y = LN2(xs)
    ln_bf16_kernel<<<4096, 256, 0, stream>>>(xs, ln2_g + i * 512, ln2_b + i * 512, ybf);
    // h = gelu(y @ ff1_w^T + ff1_b)
    gemm_kernel<1><<<dim3(8, 256), 256, 0, stream>>>(ybf, 512, w1 + (size_t)i * 2048 * 512, 512,
                                                     ff1_b + i * 2048, nullptr, nullptr, big, 2048);
    // out = xs + h @ ff2_w^T + ff2_b   (last layer writes d_out)
    float* Rout = (i == 1) ? (float*)d_out : xs;
    gemm_kernel<2><<<dim3(2, 256), 256, 0, stream>>>(big, 2048, w2 + (size_t)i * 512 * 2048, 2048,
                                                     ff2_b + i * 512, xs, Rout, nullptr, 512);
  }
}
